// HungarianMatcher_44951127720440
// MI455X (gfx1250) — matrix operand order repaired and checked by reading, not compiled, not run on hardware
//
#include <hip/hip_runtime.h>
#include <hip/hip_bf16.h>
#include <math.h>

typedef __attribute__((ext_vector_type(16))) _Float16 v16h;
typedef __attribute__((ext_vector_type(8)))  float    v8f;

#define T_DIM 4096
#define N_DIM 128
#define L_DIM 64
#define EPSF  1e-6f

// Per-element transform of a logit x (one expf, one log1pf, four sqrtf):
//  p    = sigmoid(x)
//  diff = (1-p+eps)^.25 * softplus(-x)  -  (p+eps)^.25 * softplus(x)   [t=1 minus t=0 branch]
//  Bv   = (p+eps)^.25 * softplus(x)                                    [t=0 branch]
__device__ __forceinline__ void elem_transform(float x, float& diff, float& p, float& Bv) {
    const float ax  = fabsf(x);
    const float z   = expf(-ax);               // e^{-|x|}
    const float inv = 1.0f / (1.0f + z);       // sigmoid(|x|)
    const float s   = (x >= 0.0f) ? inv : (1.0f - inv);
    const float sp_neg = fmaxf(-x, 0.0f) + log1pf(z);   // softplus(-x)
    const float sp_pos = sp_neg + x;                    // softplus(x)
    const float fw1 = sqrtf(sqrtf(1.0f - s + EPSF));    // y^0.25
    const float fw0 = sqrtf(sqrtf(s + EPSF));
    Bv   = fw0 * sp_pos;
    diff = fw1 * sp_neg - Bv;
    p    = s;
}

// ---- Pack binary targets once into f16 B-fragment order; emit sumT partials ----
// tpack[((b*128 + kt)*4 + mt)*32 + lane][0..15]
//   lane col = m0 + (lane&15); K = 2*(g&3)+e + (g>=4?16:0) + (lane>=16?8:0).
__global__ void __launch_bounds__(32) pack_targets_kernel(const float* __restrict__ targets,
                                                          _Float16* __restrict__ tpack,
                                                          float* __restrict__ sumT_part) {
    const int lane = threadIdx.x;
    const int mt   = blockIdx.x & 3;
    const int kt   = (blockIdx.x >> 2) & 127;
    const int b    = blockIdx.x >> 9;
    const int row  = lane & 15;
    const int kBhi = (lane >> 4) * 8;
    const int k0   = kt << 5;
    const int m    = (mt << 4) + row;
    const float* tg = targets + (size_t)b * T_DIM * L_DIM;
    _Float16* dst = tpack + ((size_t)blockIdx.x * 32 + lane) * 16;

    float sT = 0.0f;
    v16h frag;
    #pragma unroll
    for (int g = 0; g < 8; ++g)
        #pragma unroll
        for (int e = 0; e < 2; ++e) {
            const int k = ((g & 3) << 1) + e + ((g & 4) ? 16 : 0) + kBhi;
            const float v = tg[(size_t)(k0 + k) * L_DIM + m];
            sT += v;
            frag[2 * g + e] = (_Float16)v;
        }
    *(v16h*)dst = frag;
    sT += __shfl_down(sT, 16, 32);             // combine lower/upper K halves of column m
    if (lane < 16) sumT_part[(size_t)blockIdx.x * 16 + row] = sT;
}

// ---- Wide pre-pass: elementwise transform -> packed f16 A fragments + sumB/sumP partials ----
// apack[(((b*8 + nt)*128 + kt)*4 + mat)*32 + lane][0..15], mat: 0=dh 1=dl 2=ph 3=pl
// A f16 16x32 layout: lane row = lane&15; K = 2*(g&3)+e + (g>=4?16:0) + (lane>=16?8:0).
__global__ void __launch_bounds__(32) prepass_kernel(const float* __restrict__ preds_mask,
                                                     _Float16* __restrict__ apack,
                                                     float* __restrict__ sumB_part,
                                                     float* __restrict__ sumP_part) {
    const int lane = threadIdx.x;
    const int kt   = blockIdx.x & 127;
    const int nt   = (blockIdx.x >> 7) & 7;
    const int b    = blockIdx.x >> 10;
    const int row  = lane & 15;
    const int kAhi = (lane >> 4) * 8;
    const int k0   = kt << 5;
    const int n    = (nt << 4) + row;
    const float* pm = preds_mask + (size_t)b * T_DIM * N_DIM;
    _Float16* base = apack + (((size_t)blockIdx.x * 4) * 32 + lane) * 16;

    float sB = 0.0f, sP = 0.0f;
    v16h a_dh, a_dl, a_ph, a_pl;
    #pragma unroll
    for (int g = 0; g < 8; ++g)
        #pragma unroll
        for (int e = 0; e < 2; ++e) {
            const int k = ((g & 3) << 1) + e + ((g & 4) ? 16 : 0) + kAhi;
            float d, p, Bv;
            elem_transform(pm[(size_t)(k0 + k) * N_DIM + n], d, p, Bv);
            sB += Bv; sP += p;
            const _Float16 dh = (_Float16)d;
            const _Float16 ph = (_Float16)p;
            a_dh[2 * g + e] = dh;
            a_dl[2 * g + e] = (_Float16)(d - (float)dh);
            a_ph[2 * g + e] = ph;
            a_pl[2 * g + e] = (_Float16)(p - (float)ph);
        }
    *(v16h*)(base +    0) = a_dh;   // mat stride = 32 lanes * 16 elems = 512 f16
    *(v16h*)(base +  512) = a_dl;
    *(v16h*)(base + 1024) = a_ph;
    *(v16h*)(base + 1536) = a_pl;

    sB += __shfl_down(sB, 16, 32);             // combine the two K halves of row n
    sP += __shfl_down(sP, 16, 32);
    if (lane < 16) {
        sumB_part[(size_t)blockIdx.x * 16 + row] = sB;
        sumP_part[(size_t)blockIdx.x * 16 + row] = sP;
    }
}

// ---- Reduce partials over the 128 k-tiles (fixed order -> deterministic) ----
__global__ void __launch_bounds__(32) reduce_sums_kernel(const float* __restrict__ sumB_part,
                                                         const float* __restrict__ sumP_part,
                                                         const float* __restrict__ sumT_part,
                                                         float* __restrict__ sumB,
                                                         float* __restrict__ sumP,
                                                         float* __restrict__ sumT) {
    const int wid  = blockIdx.x;
    const int lane = threadIdx.x;
    if (wid < 2 * N_DIM) {                       // sumB[n], sumP[n]
        const int b = wid >> 7;
        const int n = wid & (N_DIM - 1);
        const int nt = n >> 4, r = n & 15;
        float sB = 0.0f, sP = 0.0f;
        for (int kt = lane; kt < 128; kt += 32) {
            const size_t idx = ((size_t)((b * 8 + nt) * 128 + kt)) * 16 + r;
            sB += sumB_part[idx];
            sP += sumP_part[idx];
        }
        #pragma unroll
        for (int off = 16; off > 0; off >>= 1) {
            sB += __shfl_down(sB, off, 32);
            sP += __shfl_down(sP, off, 32);
        }
        if (lane == 0) { sumB[wid] = sB; sumP[wid] = sP; }
    } else {                                     // sumT[m]
        const int id = wid - 2 * N_DIM;
        const int b  = id >> 6;
        const int m  = id & (L_DIM - 1);
        const int mt = m >> 4, r = m & 15;
        float sT = 0.0f;
        for (int kt = lane; kt < 128; kt += 32)
            sT += sumT_part[((size_t)((b * 128 + kt) * 4 + mt)) * 16 + r];
        #pragma unroll
        for (int off = 16; off > 0; off >>= 1) sT += __shfl_down(sT, off, 32);
        if (lane == 0) sumT[id] = sT;
    }
}

// ---- WMMA GEMM: pure load->wmma streaming; grid = B*8 n-tiles, 8 waves split K ----
__global__ void __launch_bounds__(256) gemm_kernel(const _Float16* __restrict__ apack,
                                                   const _Float16* __restrict__ tpack,
                                                   float* __restrict__ dia_dot,
                                                   float* __restrict__ tp_out) {
    const int tid   = threadIdx.x;
    const int lane  = tid & 31;
    const int wave  = tid >> 5;                 // 0..7 -> K chunk
    const int b     = blockIdx.x >> 3;
    const int ntile = blockIdx.x & 7;
    const int n0    = ntile << 4;
    const int row   = lane & 15;
    const int hi    = lane >> 4;

    v8f acc_dia[4], acc_tp[4];
    #pragma unroll
    for (int mt = 0; mt < 4; ++mt)
        #pragma unroll
        for (int r = 0; r < 8; ++r) { acc_dia[mt][r] = 0.0f; acc_tp[mt][r] = 0.0f; }

    const int ktstart = wave * 16;              // 512/32 = 16 k-tiles per wave
    for (int kt = ktstart; kt < ktstart + 16; ++kt) {
        const _Float16* apk = apack + (((size_t)((b * 8 + ntile) * 128 + kt) * 4) * 32 + lane) * 16;
        const v16h a_dh = *(const v16h*)(apk +    0);
        const v16h a_dl = *(const v16h*)(apk +  512);
        const v16h a_ph = *(const v16h*)(apk + 1024);
        const v16h a_pl = *(const v16h*)(apk + 1536);
        const _Float16* tpk = tpack + (((size_t)(b * 128 + kt) * 4) * 32 + lane) * 16;
        #pragma unroll
        for (int mt = 0; mt < 4; ++mt) {
            const v16h bf = *(const v16h*)(tpk + (size_t)mt * 512);
            acc_dia[mt] = __builtin_amdgcn_wmma_f32_16x16x32_f16(false, a_dh, false, bf, (short)0, acc_dia[mt], false, false);
            acc_dia[mt] = __builtin_amdgcn_wmma_f32_16x16x32_f16(false, a_dl, false, bf, (short)0, acc_dia[mt], false, false);
            acc_tp[mt]  = __builtin_amdgcn_wmma_f32_16x16x32_f16(false, a_ph, false, bf, (short)0, acc_tp[mt],  false, false);
            acc_tp[mt]  = __builtin_amdgcn_wmma_f32_16x16x32_f16(false, a_pl, false, bf, (short)0, acc_tp[mt],  false, false);
        }
    }

    // Fixed-order cross-wave reduction via LDS (deterministic; no atomics)
    __shared__ float red[8][8][256];   // [wave][mt*2+mat][elem] = 64 KB
    #pragma unroll
    for (int mt = 0; mt < 4; ++mt)
        #pragma unroll
        for (int r = 0; r < 8; ++r) {
            const int e = (r + (hi << 3)) * 16 + row;   // C layout: M = r + 8*hi, Ncol = row
            red[wave][mt * 2 + 0][e] = acc_dia[mt][r];
            red[wave][mt * 2 + 1][e] = acc_tp[mt][r];
        }
    __syncthreads();

    for (int flat = tid; flat < 2048; flat += 256) {
        const int slot = flat >> 8;
        const int elem = flat & 255;
        float s = 0.0f;
        #pragma unroll
        for (int w = 0; w < 8; ++w) s += red[w][slot][elem];   // fixed order
        const int mt  = slot >> 1;
        const int mat = slot & 1;
        const int rr  = elem >> 4;
        const int cc  = elem & 15;
        const size_t gi = ((size_t)b * N_DIM + n0 + rr) * L_DIM + (mt << 4) + cc;
        if (mat == 0) dia_dot[gi] = s; else tp_out[gi] = s;
    }
}

// ---- Finalize: tversky nonlinearity, focal scaling, class cost ----
__global__ void __launch_bounds__(256) finalize_kernel(const float* __restrict__ dia_dot,
                                                       const float* __restrict__ tp_in,
                                                       const float* __restrict__ sumB,
                                                       const float* __restrict__ sumP,
                                                       const float* __restrict__ sumT,
                                                       const float* __restrict__ preds_class,
                                                       const float* __restrict__ weight,
                                                       const float* __restrict__ alpha,
                                                       const float* __restrict__ beta,
                                                       float* __restrict__ out) {
    const int i = blockIdx.x * blockDim.x + threadIdx.x;
    if (i >= 2 * N_DIM * L_DIM) return;
    const int m = i & (L_DIM - 1);
    const int n = (i >> 6) & (N_DIM - 1);
    const int b = i >> 13;

    float wsum = 0.0f;
    #pragma unroll 8
    for (int j = 0; j < L_DIM; ++j) wsum += weight[j];
    const float wn = weight[m] / wsum;

    // focal: w_m * (dot(diff,t) + sum Bv) / T
    const float cost_dia = wn * (dia_dot[i] + sumB[b * N_DIM + n]) * (1.0f / (float)T_DIM);

    // tversky
    const float tp = tp_in[i];
    const float fp = sumP[b * N_DIM + n] - tp;
    const float fn = sumT[b * L_DIM + m] - tp;
    const float tv = (tp + EPSF) / (tp + alpha[m] * fp + beta[m] * fn + EPSF);
    const float y  = 1.0f - tv + EPSF;
    const float sy = sqrtf(y);
    const float cost_dice = sy * sqrtf(sy);            // y^0.75

    const float cls = preds_class[b * N_DIM + n];
    const float cost_cls = -1.0f / (1.0f + expf(-cls));

    out[i] = 5.0f * cost_dia + 5.0f * cost_dice + 2.0f * cost_cls;
}

extern "C" void kernel_launch(void* const* d_in, const int* in_sizes, int n_in,
                              void* d_out, int out_size, void* d_ws, size_t ws_size,
                              hipStream_t stream) {
    const float* preds_mask  = (const float*)d_in[0];
    const float* preds_class = (const float*)d_in[1];
    const float* targets     = (const float*)d_in[2];
    const float* weight      = (const float*)d_in[3];
    const float* alpha       = (const float*)d_in[4];
    const float* beta        = (const float*)d_in[5];
    float* out = (float*)d_out;

    float* ws        = (float*)d_ws;
    float* dia_dot   = ws;                 // 16384 floats
    float* tp        = ws + 16384;         // 16384
    float* sumB      = ws + 32768;         // 256
    float* sumP      = ws + 33024;         // 256
    float* sumT      = ws + 33280;         // 128
    float* sumB_part = ws + 33408;         // 2*8*128*16 = 32768
    float* sumP_part = ws + 66176;         // 32768
    float* sumT_part = ws + 98944;         // 2*128*4*16 = 16384
    _Float16* tpack  = (_Float16*)(ws + 115328);   // 2*128*4*32*16 f16 = 1 MB
    _Float16* apack  = (_Float16*)(ws + 377472);   // 2*8*128*4*32*16 f16 = 8 MB
    // total ~9.5 MB; every cell written before read

    pack_targets_kernel<<<2 * 128 * 4, 32, 0, stream>>>(targets, tpack, sumT_part);
    prepass_kernel<<<2 * 8 * 128, 32, 0, stream>>>(preds_mask, apack, sumB_part, sumP_part);
    reduce_sums_kernel<<<2 * N_DIM + 2 * L_DIM, 32, 0, stream>>>(sumB_part, sumP_part, sumT_part,
                                                                 sumB, sumP, sumT);
    gemm_kernel<<<16, 256, 0, stream>>>(apack, tpack, dia_dot, tp);
    finalize_kernel<<<64, 256, 0, stream>>>(dia_dot, tp, sumB, sumP, sumT,
                                            preds_class, weight, alpha, beta, out);
}
